// SelfAttentionBlock_24352464569698
// MI455X (gfx1250) — matrix operand order repaired and checked by reading, not compiled, not run on hardware
//
#include <hip/hip_runtime.h>

#define DIM 1024
#define NHEADS 16
#define HD 64
#define SEQ 2048
#define BATCH 2
#define MLP_HID 4096

typedef __attribute__((ext_vector_type(16))) __bf16        v16bf;
typedef __attribute__((ext_vector_type(8)))  float         v8f;
typedef __attribute__((ext_vector_type(8)))  unsigned int  v8u;

typedef int v4i __attribute__((vector_size(16)));
typedef __attribute__((address_space(1))) v4i* gptr_v4i;   // global (AS1) 16B vector ptr
typedef __attribute__((address_space(3))) v4i* lptr_v4i;   // LDS (AS3) 16B vector ptr

#if __has_builtin(__builtin_amdgcn_global_load_async_to_lds_b128) && \
    __has_builtin(__builtin_amdgcn_s_wait_asynccnt)
#define HAVE_ASYNC_LDS 1
#else
#define HAVE_ASYNC_LDS 0
#endif

__device__ __forceinline__ unsigned short f32_bf16(float x) {
  unsigned int u = __float_as_uint(x);
  u += 0x7FFFu + ((u >> 16) & 1u);          // round-to-nearest-even
  return (unsigned short)(u >> 16);
}
__device__ __forceinline__ unsigned int pack2(float a, float b) {
  return (unsigned int)f32_bf16(a) | ((unsigned int)f32_bf16(b) << 16);
}
__device__ __forceinline__ v8f wmma_bf16(v16bf a, v16bf b, v8f c) {
  return __builtin_amdgcn_wmma_f32_16x16x32_bf16(false, a, false, b, (short)0, c, false, false);
}

// 16B global -> LDS copy; async (ASYNCcnt-tracked) path on gfx1250
__device__ __forceinline__ void cp_lds_b128(void* lds_dst, const void* gsrc) {
#if HAVE_ASYNC_LDS
  __builtin_amdgcn_global_load_async_to_lds_b128((gptr_v4i)gsrc, (lptr_v4i)lds_dst, 0, 0);
#else
  *reinterpret_cast<uint4*>(lds_dst) = *reinterpret_cast<const uint4*>(gsrc);
#endif
}
__device__ __forceinline__ void wait_cp() {
#if HAVE_ASYNC_LDS
  __builtin_amdgcn_s_wait_asynccnt(0);
#endif
}

// A-matrix fragment (16x32 bf16, ISA 7.12.2): lane group g holds K = {col0+g*8..+7, col0+16+g*8..+7}
__device__ __forceinline__ v16bf ld_fragA(const unsigned short* base, int row0, int ld,
                                          int col0, int laneM, int g) {
  const unsigned short* p = base + (row0 + laneM) * ld + col0 + g * 8;
  uint4 lo = *reinterpret_cast<const uint4*>(p);
  uint4 hi = *reinterpret_cast<const uint4*>(p + 16);
  v8u u; u[0]=lo.x; u[1]=lo.y; u[2]=lo.z; u[3]=lo.w; u[4]=hi.x; u[5]=hi.y; u[6]=hi.z; u[7]=hi.w;
  return __builtin_bit_cast(v16bf, u);
}
// B-matrix fragment (32x16 bf16), LDS holds B^T = [N][K]: lane group g holds K = {col0+g*8..+7, col0+16+g*8..+7}
__device__ __forceinline__ v16bf ld_fragB(const unsigned short* base, int row0, int ld,
                                          int col0, int laneM, int g) {
  const unsigned short* p = base + (row0 + laneM) * ld + col0 + g * 8;
  uint4 lo = *reinterpret_cast<const uint4*>(p);
  uint4 hi = *reinterpret_cast<const uint4*>(p + 16);
  v8u u; u[0]=lo.x; u[1]=lo.y; u[2]=lo.z; u[3]=lo.w; u[4]=hi.x; u[5]=hi.y; u[6]=hi.z; u[7]=hi.w;
  return __builtin_bit_cast(v16bf, u);
}

__device__ __forceinline__ float gelu_f(float x) {   // tanh approx (jax.nn.gelu default)
  float x3 = x * x * x;
  float t = tanhf(0.7978845608028654f * (x + 0.044715f * x3));
  return 0.5f * x * (1.0f + t);
}

// --------------------------------------------------- precision prep kernels ---
__global__ __launch_bounds__(256) void convert_bf16_kernel(
    const float* __restrict__ src, unsigned short* __restrict__ dst, int n4)
{
  int i = blockIdx.x * blockDim.x + threadIdx.x;
  if (i < n4) {
    float4 f = reinterpret_cast<const float4*>(src)[i];
    uint2 u; u.x = pack2(f.x, f.y); u.y = pack2(f.z, f.w);
    reinterpret_cast<uint2*>(dst)[i] = u;
  }
}

// Wt[n][k] (bf16) = W[k][n] (f32); 16x16 LDS tile transpose, coalesced both sides
__global__ __launch_bounds__(256) void transpose_w_kernel(
    const float* __restrict__ W, unsigned short* __restrict__ Wt, int K, int N)
{
  __shared__ float tile[16][17];
  const int tx = threadIdx.x & 15, ty = threadIdx.x >> 4;
  const int k0 = blockIdx.y * 16, n0 = blockIdx.x * 16;
  tile[ty][tx] = W[(size_t)(k0 + ty) * N + n0 + tx];
  __syncthreads();
  Wt[(size_t)(n0 + ty) * K + k0 + tx] = f32_bf16(tile[tx][ty]);
}

// ---------------------------------------------------------------- GEMM --------
// C[M,N] = act(A[M,K] @ Wt[N,K]^T + bias[N]) (+ res[M,N]); A,Wt bf16; out f32 and/or bf16
#define BM 128
#define BN 128
#define BK 32
#define LDS_T 56   // padded bf16 stride: 112B rows, 16B-aligned b128 frags, conflict-free

__global__ __launch_bounds__(256) void gemm_bf16_kernel(
    const unsigned short* __restrict__ Ab, const unsigned short* __restrict__ Wt,
    const float* __restrict__ bias, const float* __restrict__ res,
    float* __restrict__ Cf, unsigned short* __restrict__ Cb,
    int N, int K, int act)
{
  __shared__ unsigned short sA[2][BM * LDS_T];
  __shared__ unsigned short sB[2][BN * LDS_T];

  const int tid   = threadIdx.x;
  const int lane  = tid & 31;
  const int laneM = lane & 15;
  const int g     = lane >> 4;
  const int wid   = tid >> 5;
  const int wm    = (wid >> 1) * 32;   // wave tile: 32 x 64
  const int wn    = (wid & 1) * 64;
  const int blk_m = blockIdx.y * BM;
  const int blk_n = blockIdx.x * BN;

  const int srow = tid >> 1;           // staging: 128 rows x 32 cols, 2 threads/row
  const int sc0  = (tid & 1) * 16;

  v8f acc[2][4];
  #pragma unroll
  for (int i = 0; i < 2; ++i)
    #pragma unroll
    for (int j = 0; j < 4; ++j) acc[i][j] = (v8f){0,0,0,0,0,0,0,0};

  auto stage = [&](int buf, int k0) {
    const unsigned short* a = Ab + (size_t)(blk_m + srow) * K + k0 + sc0;
    cp_lds_b128(&sA[buf][srow * LDS_T + sc0],     a);
    cp_lds_b128(&sA[buf][srow * LDS_T + sc0 + 8], a + 8);
    const unsigned short* w = Wt + (size_t)(blk_n + srow) * K + k0 + sc0;
    cp_lds_b128(&sB[buf][srow * LDS_T + sc0],     w);
    cp_lds_b128(&sB[buf][srow * LDS_T + sc0 + 8], w + 8);
  };

  stage(0, 0);
  const int nk = K / BK;
  for (int it = 0; it < nk; ++it) {
    wait_cp();
    __syncthreads();                      // staged data visible; prev compute done
    if (it + 1 < nk) stage((it + 1) & 1, (it + 1) * BK);
    const int buf = it & 1;
    v16bf a0 = ld_fragA(sA[buf], wm,      LDS_T, 0, laneM, g);
    v16bf a1 = ld_fragA(sA[buf], wm + 16, LDS_T, 0, laneM, g);
    #pragma unroll
    for (int tn = 0; tn < 4; ++tn) {
      v16bf bfr = ld_fragB(sB[buf], wn + tn * 16, LDS_T, 0, laneM, g);
      acc[0][tn] = wmma_bf16(a0, bfr, acc[0][tn]);
      acc[1][tn] = wmma_bf16(a1, bfr, acc[1][tn]);
    }
  }

  #pragma unroll
  for (int tm = 0; tm < 2; ++tm)
    #pragma unroll
    for (int tn = 0; tn < 4; ++tn) {
      const int gn = blk_n + wn + tn * 16 + laneM;
      #pragma unroll
      for (int e = 0; e < 8; ++e) {       // C/D layout: VGPR e -> row e + 8*(lane/16)
        const int gm = blk_m + wm + tm * 16 + e + 8 * g;
        float val = acc[tm][tn][e] + bias[gn];
        if (act) val = gelu_f(val);
        if (res) val += res[(size_t)gm * N + gn];
        const size_t idx = (size_t)gm * N + gn;
        if (Cf) Cf[idx] = val;
        if (Cb) Cb[idx] = f32_bf16(val);
      }
    }
}

// ------------------------------------------------------- Flash attention ------
#define LDK_S 72   // padded bf16 stride (144B rows, 16B aligned)

__global__ __launch_bounds__(128) void flash_attn_kernel(
    const unsigned short* __restrict__ qp, const unsigned short* __restrict__ kp,
    const unsigned short* __restrict__ vp, const float* __restrict__ mask,
    unsigned short* __restrict__ ctx)
{
  __shared__ unsigned short sK[64 * LDK_S];       // [key][dim]
  __shared__ unsigned short sV[64 * LDK_S];       // [dim][key] (transposed)
  __shared__ unsigned short sP[4 * 16 * LDK_S];   // per-wave P tiles [row][key]

  const int tid   = threadIdx.x;
  const int lane  = tid & 31;
  const int laneM = lane & 15;
  const int g     = lane >> 4;
  const int wid   = tid >> 5;
  const int b  = blockIdx.z;
  const int h  = blockIdx.y;
  const int q0 = blockIdx.x * 64;

  // Q fragments for this wave's 16 rows (bf16 gather in A-fragment order)
  const unsigned short* qrow = qp + ((size_t)(b * SEQ + q0 + wid * 16 + laneM)) * DIM + h * HD;
  v16bf qf[2];
  #pragma unroll
  for (int ks = 0; ks < 2; ++ks) {
    v8u u;
    #pragma unroll
    for (int j = 0; j < 8; ++j) {
      int i0 = 2 * j, i1 = 2 * j + 1;
      int d0 = ks * 32 + ((i0 < 8) ? i0 : i0 + 8) + g * 8;
      int d1 = ks * 32 + ((i1 < 8) ? i1 : i1 + 8) + g * 8;
      u[j] = (unsigned int)qrow[d0] | ((unsigned int)qrow[d1] << 16);
    }
    qf[ks] = __builtin_bit_cast(v16bf, u);
  }

  float m_s[8], l_s[8];
  #pragma unroll
  for (int e = 0; e < 8; ++e) { m_s[e] = -3.0e38f; l_s[e] = 0.0f; }
  v8f o[4];
  #pragma unroll
  for (int t = 0; t < 4; ++t) o[t] = (v8f){0,0,0,0,0,0,0,0};

  const int skey = tid >> 1;
  const int sc0  = (tid & 1) * 32;

  for (int kv0 = 0; kv0 < SEQ; kv0 += 64) {
    __syncthreads();                       // prior reads of sK/sV done
    { // stage K tile [key][dim]: straight bf16 copy, async path when available
      const unsigned short* src = kp + ((size_t)(b * SEQ + kv0 + skey)) * DIM + h * HD + sc0;
      #pragma unroll
      for (int j = 0; j < 4; ++j)
        cp_lds_b128(&sK[skey * LDK_S + sc0 + j * 8], src + j * 8);
    }
    { // stage V tile transposed [dim][key]
      const unsigned short* src = vp + ((size_t)(b * SEQ + kv0 + skey)) * DIM + h * HD + sc0;
      #pragma unroll
      for (int d = 0; d < 32; ++d)
        sV[(sc0 + d) * LDK_S + skey] = src[d];
    }
    wait_cp();
    __syncthreads();

    // S = Q @ K^T (16 x 64 logits per wave)
    v8f s[4];
    #pragma unroll
    for (int t = 0; t < 4; ++t) s[t] = (v8f){0,0,0,0,0,0,0,0};
    #pragma unroll
    for (int tn = 0; tn < 4; ++tn)
      #pragma unroll
      for (int ks = 0; ks < 2; ++ks) {
        v16bf kf = ld_fragB(sK, tn * 16, LDK_S, ks * 32, laneM, g);
        s[tn] = wmma_bf16(qf[ks], kf, s[tn]);
      }

    // scale (1/sqrt(64)) + additive mask
    #pragma unroll
    for (int tn = 0; tn < 4; ++tn) {
      float mk = -1.0e9f * mask[(size_t)b * SEQ + kv0 + tn * 16 + laneM];
      #pragma unroll
      for (int e = 0; e < 8; ++e) s[tn][e] = s[tn][e] * 0.125f + mk;
    }

    // online softmax; rows shared across the 16 lanes with equal lane/16
    #pragma unroll
    for (int e = 0; e < 8; ++e) {
      float mx = fmaxf(fmaxf(s[0][e], s[1][e]), fmaxf(s[2][e], s[3][e]));
      mx = fmaxf(mx, __shfl_xor(mx, 1, 32));
      mx = fmaxf(mx, __shfl_xor(mx, 2, 32));
      mx = fmaxf(mx, __shfl_xor(mx, 4, 32));
      mx = fmaxf(mx, __shfl_xor(mx, 8, 32));
      float mnew = fmaxf(m_s[e], mx);
      float corr = __expf(m_s[e] - mnew);
      float rsum = 0.0f;
      #pragma unroll
      for (int tn = 0; tn < 4; ++tn) {
        float p = __expf(s[tn][e] - mnew);
        s[tn][e] = p;
        rsum += p;
      }
      rsum += __shfl_xor(rsum, 1, 32);
      rsum += __shfl_xor(rsum, 2, 32);
      rsum += __shfl_xor(rsum, 4, 32);
      rsum += __shfl_xor(rsum, 8, 32);
      l_s[e] = l_s[e] * corr + rsum;
      m_s[e] = mnew;
      #pragma unroll
      for (int tn = 0; tn < 4; ++tn) o[tn][e] *= corr;
    }

    // P: C-layout -> per-wave LDS (same-wave DS ordering) -> A-layout fragments
    unsigned short* pw = &sP[wid * 16 * LDK_S];
    #pragma unroll
    for (int tn = 0; tn < 4; ++tn)
      #pragma unroll
      for (int e = 0; e < 8; ++e)
        pw[(e + 8 * g) * LDK_S + tn * 16 + laneM] = f32_bf16(s[tn][e]);

    // O += P @ V
    #pragma unroll
    for (int ks = 0; ks < 2; ++ks) {
      v16bf pf = ld_fragA(pw, 0, LDK_S, ks * 32, laneM, g);
      #pragma unroll
      for (int dn = 0; dn < 4; ++dn) {
        v16bf vf = ld_fragB(sV, dn * 16, LDK_S, ks * 32, laneM, g);
        o[dn] = wmma_bf16(pf, vf, o[dn]);
      }
    }
  }

  // normalize and write merged-head context [B,S,DIM] (bf16 for next GEMM)
  #pragma unroll
  for (int dn = 0; dn < 4; ++dn)
    #pragma unroll
    for (int e = 0; e < 8; ++e) {
      int grow = q0 + wid * 16 + e + 8 * g;
      float val = o[dn][e] / l_s[e];
      ctx[((size_t)(b * SEQ + grow)) * DIM + h * HD + dn * 16 + laneM] = f32_bf16(val);
    }
}

// ---------------------------------------------------------------- launch ------
extern "C" void kernel_launch(void* const* d_in, const int* in_sizes, int n_in,
                              void* d_out, int out_size, void* d_ws, size_t ws_size,
                              hipStream_t stream)
{
  (void)in_sizes; (void)n_in; (void)out_size; (void)ws_size;
  const float* v    = (const float*)d_in[0];
  const float* k    = (const float*)d_in[1];
  const float* q    = (const float*)d_in[2];
  const float* mask = (const float*)d_in[3];
  const float* wq_w = (const float*)d_in[4];
  const float* wq_b = (const float*)d_in[5];
  const float* wk_w = (const float*)d_in[6];
  const float* wk_b = (const float*)d_in[7];
  const float* wv_w = (const float*)d_in[8];
  const float* wv_b = (const float*)d_in[9];
  const float* fc_w = (const float*)d_in[10];
  const float* fc_b = (const float*)d_in[11];
  const float* w1   = (const float*)d_in[12];
  const float* b1   = (const float*)d_in[13];
  const float* w2   = (const float*)d_in[14];
  const float* b2   = (const float*)d_in[15];
  float* out = (float*)d_out;

  const int M = BATCH * SEQ;                       // 4096
  typedef unsigned short ush;

  char* w    = (char*)d_ws;
  auto alloc = [&](size_t bytes) { char* p = w; w += (bytes + 255) & ~(size_t)255; return p; };
  const size_t MD2 = (size_t)M * DIM * 2;
  ush* qb    = (ush*)alloc(MD2);
  ush* kb    = (ush*)alloc(MD2);
  ush* vb    = (ush*)alloc(MD2);
  ush* wqT   = (ush*)alloc((size_t)DIM * DIM * 2);
  ush* wkT   = (ush*)alloc((size_t)DIM * DIM * 2);
  ush* wvT   = (ush*)alloc((size_t)DIM * DIM * 2);
  ush* fcT   = (ush*)alloc((size_t)DIM * DIM * 2);
  ush* w1T   = (ush*)alloc((size_t)DIM * MLP_HID * 2);
  ush* w2T   = (ush*)alloc((size_t)MLP_HID * DIM * 2);
  ush* qpb   = (ush*)alloc(MD2);
  ush* kpb   = (ush*)alloc(MD2);
  ush* vpb   = (ush*)alloc(MD2);
  ush* ctxb  = (ush*)alloc(MD2);
  float* o1f = (float*)alloc((size_t)M * DIM * 4);
  ush* o1b   = (ush*)alloc(MD2);
  ush* hidb  = (ush*)alloc((size_t)M * MLP_HID * 2);

  // --- precision prep: activations -> bf16, weights -> transposed bf16 [N][K]
  const int n4 = M * DIM / 4;
  convert_bf16_kernel<<<dim3((n4 + 255) / 256), dim3(256), 0, stream>>>(q, qb, n4);
  convert_bf16_kernel<<<dim3((n4 + 255) / 256), dim3(256), 0, stream>>>(k, kb, n4);
  convert_bf16_kernel<<<dim3((n4 + 255) / 256), dim3(256), 0, stream>>>(v, vb, n4);
  transpose_w_kernel<<<dim3(DIM / 16, DIM / 16), dim3(256), 0, stream>>>(wq_w, wqT, DIM, DIM);
  transpose_w_kernel<<<dim3(DIM / 16, DIM / 16), dim3(256), 0, stream>>>(wk_w, wkT, DIM, DIM);
  transpose_w_kernel<<<dim3(DIM / 16, DIM / 16), dim3(256), 0, stream>>>(wv_w, wvT, DIM, DIM);
  transpose_w_kernel<<<dim3(DIM / 16, DIM / 16), dim3(256), 0, stream>>>(fc_w, fcT, DIM, DIM);
  transpose_w_kernel<<<dim3(MLP_HID / 16, DIM / 16), dim3(256), 0, stream>>>(w1, w1T, DIM, MLP_HID);
  transpose_w_kernel<<<dim3(DIM / 16, MLP_HID / 16), dim3(256), 0, stream>>>(w2, w2T, MLP_HID, DIM);

  dim3 blk(256);
  dim3 gN(DIM / BN, M / BM);       // N = 1024
  dim3 gH(MLP_HID / BN, M / BM);   // N = 4096

  // --- QKV projections (bf16 out only)
  gemm_bf16_kernel<<<gN, blk, 0, stream>>>(qb, wqT, wq_b, nullptr, nullptr, qpb, DIM, DIM, 0);
  gemm_bf16_kernel<<<gN, blk, 0, stream>>>(kb, wkT, wk_b, nullptr, nullptr, kpb, DIM, DIM, 0);
  gemm_bf16_kernel<<<gN, blk, 0, stream>>>(vb, wvT, wv_b, nullptr, nullptr, vpb, DIM, DIM, 0);

  // --- attention
  flash_attn_kernel<<<dim3(SEQ / 64, NHEADS, BATCH), dim3(128), 0, stream>>>(qpb, kpb, vpb, mask, ctxb);

  // --- out proj (+residual v), MLP
  gemm_bf16_kernel<<<gN, blk, 0, stream>>>(ctxb, fcT, fc_b, v,    o1f, o1b,  DIM,     DIM,     0);
  gemm_bf16_kernel<<<gH, blk, 0, stream>>>(o1b,  w1T, b1,   nullptr, nullptr, hidb, MLP_HID, DIM, 1);
  gemm_bf16_kernel<<<gN, blk, 0, stream>>>(hidb, w2T, b2,   o1f,  out, nullptr, DIM,  MLP_HID, 0);
}
